// Model_5153960755634
// MI455X (gfx1250) — matrix operand order repaired and checked by reading, not compiled, not run on hardware
//
#include <hip/hip_runtime.h>

// ---------------------------------------------------------------------------
// Types for CDNA5 WMMA (gfx1250, wave32)
// ---------------------------------------------------------------------------
typedef __attribute__((ext_vector_type(16))) __bf16 v16bf;
typedef __attribute__((ext_vector_type(8)))  float  v8f;

union Frag {
    v16bf v;
    uint4 q[2];
};

// float -> bf16 bits, round-to-nearest-even (fallback path)
static __device__ __forceinline__ unsigned int f2bf(float f) {
    unsigned int x = __float_as_uint(f);
    return (x + 0x7FFFu + ((x >> 16) & 1u)) >> 16;
}

// pack two floats -> packed bf16x2 dword; prefer HW packed converter
#if __has_builtin(__builtin_amdgcn_cvt_pk_bf16_f32)
static __device__ __forceinline__ unsigned int pk2bf(float lo, float hi) {
    auto r = __builtin_amdgcn_cvt_pk_bf16_f32(lo, hi);  // v_cvt_pk_bf16_f32
    return __builtin_bit_cast(unsigned int, r);
}
#else
static __device__ __forceinline__ unsigned int pk2bf(float lo, float hi) {
    return (f2bf(lo) & 0xFFFFu) | (f2bf(hi) << 16);
}
#endif

// Problem constants (match reference)
#define NSOTU 100000
#define NTAXON 20000
#define NE    800000
#define NEL   200000
#define DS    256
#define DT    128
#define HD    256

#define F_ACCUM 1
#define F_RELU  2

#define APAD 40   // LDS row stride (elements) for A tile: 20 banks, 16B aligned
#define BPAD 40   // LDS row stride (elements) for B tile

// ---------------------------------------------------------------------------
// Utility kernels
// ---------------------------------------------------------------------------
__global__ void zero_f32(float* __restrict__ p, size_t n) {
    size_t i = (size_t)blockIdx.x * blockDim.x + threadIdx.x;
    size_t s = (size_t)gridDim.x * blockDim.x;
    for (; i < n; i += s) p[i] = 0.f;
}

__global__ void count_edges(float* __restrict__ cnt_s, float* __restrict__ cnt_t,
                            const int* __restrict__ src, const int* __restrict__ dst,
                            int E) {
    int i = blockIdx.x * blockDim.x + threadIdx.x;
    int s = gridDim.x * blockDim.x;
    for (; i < E; i += s) {
        atomicAdd(&cnt_s[src[i]], 1.0f);
        atomicAdd(&cnt_t[dst[i]], 1.0f);
    }
}

// accum[dstIdx[e], :] += x[srcIdx[e], :]   (one wave per edge, D floats)
__global__ void scatter_add(float* __restrict__ accum, const float* __restrict__ x,
                            const int* __restrict__ srcIdx, const int* __restrict__ dstIdx,
                            int E, int D) {
    int gwave = (blockIdx.x * blockDim.x + threadIdx.x) >> 5;
    int lane  = threadIdx.x & 31;
    int nw    = (gridDim.x * blockDim.x) >> 5;
    for (int e = gwave; e < E; e += nw) {
        int s = srcIdx[e];
        int d = dstIdx[e];
        const float* xr = x + (size_t)s * D;
        float*       ar = accum + (size_t)d * D;
        for (int j = lane; j < D; j += 32) atomicAdd(&ar[j], xr[j]);
    }
}

__global__ void divide_mean(float* __restrict__ mean, const float* __restrict__ cnt,
                            int Nn, int D) {
    size_t total = (size_t)Nn * D;
    size_t i = (size_t)blockIdx.x * blockDim.x + threadIdx.x;
    size_t s = (size_t)gridDim.x * blockDim.x;
    for (; i < total; i += s) mean[i] /= fmaxf(cnt[i / D], 1.0f);
}

// z[e, 0:H] = zs[ls[e]], z[e, H:2H] = zt[ld[e]]   (float4 vectorized, H%4==0)
__global__ void gather_concat(float4* __restrict__ z,
                              const float4* __restrict__ zs, const float4* __restrict__ zt,
                              const int* __restrict__ ls, const int* __restrict__ ld,
                              int EL, int H4 /* = H/4 */) {
    size_t total = (size_t)EL * 2 * H4;
    size_t i = (size_t)blockIdx.x * blockDim.x + threadIdx.x;
    size_t s = (size_t)gridDim.x * blockDim.x;
    for (; i < total; i += s) {
        int e = (int)(i / (2 * H4));
        int d = (int)(i % (2 * H4));
        z[i] = (d < H4) ? zs[(size_t)ls[e] * H4 + d]
                        : zt[(size_t)ld[e] * H4 + (d - H4)];
    }
}

// out[e] = dot(h[e,:], w[:]) + b   (one wave per row, float4 loads, wave32 reduce)
__global__ void decoder_out(const float4* __restrict__ h, const float4* __restrict__ w,
                            const float* __restrict__ b, float* __restrict__ out,
                            int EL, int H4 /* = H/4 */) {
    int gwave = (blockIdx.x * blockDim.x + threadIdx.x) >> 5;
    int lane  = threadIdx.x & 31;
    int nw    = (gridDim.x * blockDim.x) >> 5;
    for (int e = gwave; e < EL; e += nw) {
        const float4* hr = h + (size_t)e * H4;
        float s = 0.f;
        for (int j = lane; j < H4; j += 32) {
            float4 a = hr[j];
            float4 c = w[j];
            s += a.x * c.x + a.y * c.y + a.z * c.z + a.w * c.w;
        }
        for (int off = 16; off; off >>= 1) s += __shfl_down(s, off, 32);
        if (lane == 0) out[e] = s + b[0];
    }
}

// ---------------------------------------------------------------------------
// WMMA GEMM: C[M,N] (+)= A[M,K] @ W[N,K]^T  (+bias)(+relu)
// Block = 256 threads = 8 waves; C tile 64(M) x 128(N); each wave: 32x32
// (2x2 accumulators, 4 WMMAs / K-step). K-step 32.
//
// LDS tiles are RAW row-major (rows padded to 40 elems = 20 banks):
//  - B fragment (32x16, same k order as A at fixed n) is TWO contiguous 16B runs.
//  - A fragment (16x32, k = (e&7)+((e>>3)<<4)+(half<<3) at fixed m) is TWO
//    contiguous 16B runs: k in [8h, 8h+8) and [16+8h, 16+8h+8).
// So fragments are plain uint4 LDS loads, no swizzle math anywhere.
//
// FULL=true  : block fully interior in M and N (no guards at all).
// FULL=false : guarded tail (last partial M block).
// ---------------------------------------------------------------------------
template <bool FULL>
__global__ void gemm_wmma_bf16(const float* __restrict__ A, const float* __restrict__ W,
                               const float* __restrict__ bias, float* __restrict__ C,
                               int M, int N, int K, int flags, int mbase) {
    __shared__ __align__(16) unsigned short As[64 * APAD];   // [m][k] row-major
    __shared__ __align__(16) unsigned short Bs[128 * BPAD];  // [n][k] row-major

    const int tid   = threadIdx.x;
    const int lane  = tid & 31;
    const int wave  = tid >> 5;
    const int waveM = wave >> 2;   // 0..1 -> rows  waveM*32
    const int waveN = wave & 3;    // 0..3 -> cols  waveN*32
    const int half  = lane >> 4;
    const int l16   = lane & 15;

    const int m0 = mbase + blockIdx.y * 64;
    const int n0 = blockIdx.x * 128;

    v8f acc[2][2] = {};

    for (int kk = 0; kk < K; kk += 32) {
        // prefetch next A panel (global_prefetch_b8)
        if (kk + 32 < K) {
            int rowp = m0 + (tid >> 2);
            if (FULL || rowp < M) __builtin_prefetch(&A[(size_t)rowp * K + kk + 32], 0, 1);
        }
        // ---- stage A panel 64x32: 8 lanes cover one row (coalesced float4)
#pragma unroll
        for (int i = tid; i < 64 * 8; i += 256) {   // 2 iterations
            int m = i >> 3;
            int k = (i & 7) << 2;
            float4 v;
            if (FULL) {
                v = *(const float4*)&A[(size_t)(m0 + m) * K + kk + k];
            } else {
                v = make_float4(0.f, 0.f, 0.f, 0.f);
                if (m0 + m < M) v = *(const float4*)&A[(size_t)(m0 + m) * K + kk + k];
            }
            unsigned int* dst = (unsigned int*)&As[m * APAD + k];
            dst[0] = pk2bf(v.x, v.y);
            dst[1] = pk2bf(v.z, v.w);
        }
        // ---- stage B panel 128x32 from W[N,K]: row n holds W[n0+n][kk..kk+32)
#pragma unroll
        for (int i = tid; i < 128 * 8; i += 256) {  // 4 iterations
            int n = i >> 3;
            int k = (i & 7) << 2;
            float4 v;
            if (FULL) {
                v = *(const float4*)&W[(size_t)(n0 + n) * K + kk + k];
            } else {
                v = make_float4(0.f, 0.f, 0.f, 0.f);
                if (n0 + n < N) v = *(const float4*)&W[(size_t)(n0 + n) * K + kk + k];
            }
            unsigned int* dst = (unsigned int*)&Bs[n * BPAD + k];
            dst[0] = pk2bf(v.x, v.y);
            dst[1] = pk2bf(v.z, v.w);
        }
        __syncthreads();

        // ---- fragments: contiguous uint4 LDS reads
        Frag a0, a1, b0, b1;
        {
            const unsigned short* ar0 = &As[((waveM * 2 + 0) * 16 + l16) * APAD];
            const unsigned short* ar1 = &As[((waveM * 2 + 1) * 16 + l16) * APAD];
            a0.q[0] = *(const uint4*)&ar0[half * 8];
            a0.q[1] = *(const uint4*)&ar0[16 + half * 8];
            a1.q[0] = *(const uint4*)&ar1[half * 8];
            a1.q[1] = *(const uint4*)&ar1[16 + half * 8];
            const unsigned short* br0 = &Bs[((waveN * 2 + 0) * 16 + l16) * BPAD + half * 8];
            const unsigned short* br1 = &Bs[((waveN * 2 + 1) * 16 + l16) * BPAD + half * 8];
            b0.q[0] = *(const uint4*)&br0[0];
            b0.q[1] = *(const uint4*)&br0[16];
            b1.q[0] = *(const uint4*)&br1[0];
            b1.q[1] = *(const uint4*)&br1[16];
        }

        acc[0][0] = __builtin_amdgcn_wmma_f32_16x16x32_bf16(false, a0.v, false, b0.v,
                        (short)0, acc[0][0], false, false);
        acc[0][1] = __builtin_amdgcn_wmma_f32_16x16x32_bf16(false, a0.v, false, b1.v,
                        (short)0, acc[0][1], false, false);
        acc[1][0] = __builtin_amdgcn_wmma_f32_16x16x32_bf16(false, a1.v, false, b0.v,
                        (short)0, acc[1][0], false, false);
        acc[1][1] = __builtin_amdgcn_wmma_f32_16x16x32_bf16(false, a1.v, false, b1.v,
                        (short)0, acc[1][1], false, false);
        __syncthreads();
    }

    // ---- epilogue: lane (half,l16) holds rows r+8*half, col l16 of each tile
#pragma unroll
    for (int i = 0; i < 2; ++i) {
#pragma unroll
        for (int j = 0; j < 2; ++j) {
#pragma unroll
            for (int r = 0; r < 8; ++r) {
                int m = m0 + (waveM * 2 + i) * 16 + half * 8 + r;
                int n = n0 + (waveN * 2 + j) * 16 + l16;
                if (FULL || (m < M && n < N)) {
                    float v = acc[i][j][r];
                    if (flags & F_ACCUM) v += C[(size_t)m * N + n];
                    if (bias) v += bias[n];
                    if (flags & F_RELU) v = fmaxf(v, 0.f);
                    C[(size_t)m * N + n] = v;
                }
            }
        }
    }
}

// ---------------------------------------------------------------------------
// Launch helpers
// ---------------------------------------------------------------------------
static inline void launch_zero(float* p, size_t n, hipStream_t s) {
    int blocks = (int)(((n + 255) / 256) > 4096 ? 4096 : ((n + 255) / 256));
    if (blocks < 1) blocks = 1;
    zero_f32<<<blocks, 256, 0, s>>>(p, n);
}

// N must be a multiple of 128 (true for all calls here: N == 256).
static inline void launch_gemm(const float* A, const float* W, const float* bias,
                               float* C, int M, int N, int K, int flags,
                               hipStream_t s) {
    int gx = (N + 127) / 128;
    int fullY = M >> 6;             // fully-interior 64-row blocks
    if (fullY > 0) {
        gemm_wmma_bf16<true><<<dim3(gx, fullY), 256, 0, s>>>(A, W, bias, C,
                                                             M, N, K, flags, 0);
    }
    if (M & 63) {
        gemm_wmma_bf16<false><<<dim3(gx, 1), 256, 0, s>>>(A, W, bias, C,
                                                          M, N, K, flags, fullY * 64);
    }
}

// ---------------------------------------------------------------------------
extern "C" void kernel_launch(void* const* d_in, const int* in_sizes, int n_in,
                              void* d_out, int out_size, void* d_ws, size_t ws_size,
                              hipStream_t stream) {
    (void)in_sizes; (void)n_in; (void)out_size; (void)ws_size;

    const float* x_sotu  = (const float*)d_in[0];
    const float* x_taxon = (const float*)d_in[1];
    const int* edge_src  = (const int*)d_in[2];
    const int* edge_dst  = (const int*)d_in[3];
    const int* label_src = (const int*)d_in[4];
    const int* label_dst = (const int*)d_in[5];
    const float* Wl1_st = (const float*)d_in[6];
    const float* bl1_st = (const float*)d_in[7];
    const float* Wr1_st = (const float*)d_in[8];
    const float* Wl1_ts = (const float*)d_in[9];
    const float* bl1_ts = (const float*)d_in[10];
    const float* Wr1_ts = (const float*)d_in[11];
    const float* Wl2_st = (const float*)d_in[12];
    const float* bl2_st = (const float*)d_in[13];
    const float* Wr2_st = (const float*)d_in[14];
    const float* Wl2_ts = (const float*)d_in[15];
    const float* bl2_ts = (const float*)d_in[16];
    const float* Wr2_ts = (const float*)d_in[17];
    const float* Wlin_s = (const float*)d_in[18];
    const float* blin_s = (const float*)d_in[19];
    const float* Wlin_t = (const float*)d_in[20];
    const float* blin_t = (const float*)d_in[21];
    const float* Wd1 = (const float*)d_in[22];
    const float* bd1 = (const float*)d_in[23];
    const float* Wd2 = (const float*)d_in[24];
    const float* bd2 = (const float*)d_in[25];
    float* out = (float*)d_out;

    // workspace partition (floats)
    float* ws = (float*)d_ws;
    size_t o = 0;
    float* mean_s = ws + o; o += (size_t)NSOTU * HD;   // layer1 uses 128 cols, layer2 256
    float* mean_t = ws + o; o += (size_t)NTAXON * HD;
    float* cnt_s  = ws + o; o += (size_t)NSOTU;
    float* cnt_t  = ws + o; o += (size_t)NTAXON;
    float* h1_s   = ws + o; o += (size_t)NSOTU * HD;
    float* h1_t   = ws + o; o += (size_t)NTAXON * HD;
    float* h2_s   = ws + o; o += (size_t)NSOTU * HD;
    float* h2_t   = ws + o; o += (size_t)NTAXON * HD;
    float* zcat   = ws + o; o += (size_t)NEL * 2 * HD;
    float* hdec   = ws + o; o += (size_t)NEL * HD;
    float* z_s = h1_s;  // reuse: h1 dead after layer-2 GEMMs
    float* z_t = h1_t;

    // ---- degree counts (shared by both layers) ----
    launch_zero(cnt_s, NSOTU, stream);
    launch_zero(cnt_t, NTAXON, stream);
    count_edges<<<2048, 256, 0, stream>>>(cnt_s, cnt_t, edge_src, edge_dst, NE);

    // ---- layer 1 aggregation ----
    launch_zero(mean_t, (size_t)NTAXON * DS, stream);
    launch_zero(mean_s, (size_t)NSOTU * DT, stream);
    scatter_add<<<4096, 256, 0, stream>>>(mean_t, x_sotu, edge_src, edge_dst, NE, DS);
    scatter_add<<<4096, 256, 0, stream>>>(mean_s, x_taxon, edge_dst, edge_src, NE, DT);
    divide_mean<<<4096, 256, 0, stream>>>(mean_t, cnt_t, NTAXON, DS);
    divide_mean<<<4096, 256, 0, stream>>>(mean_s, cnt_s, NSOTU, DT);

    // ---- layer 1 GEMMs: h = relu(mean @ Wl^T + bl + x @ Wr^T) ----
    launch_gemm(mean_t, Wl1_st, nullptr, h1_t, NTAXON, HD, DS, 0, stream);
    launch_gemm(x_taxon, Wr1_st, bl1_st, h1_t, NTAXON, HD, DT, F_ACCUM | F_RELU, stream);
    launch_gemm(mean_s, Wl1_ts, nullptr, h1_s, NSOTU, HD, DT, 0, stream);
    launch_gemm(x_sotu, Wr1_ts, bl1_ts, h1_s, NSOTU, HD, DS, F_ACCUM | F_RELU, stream);

    // ---- layer 2 aggregation ----
    launch_zero(mean_t, (size_t)NTAXON * HD, stream);
    launch_zero(mean_s, (size_t)NSOTU * HD, stream);
    scatter_add<<<4096, 256, 0, stream>>>(mean_t, h1_s, edge_src, edge_dst, NE, HD);
    scatter_add<<<4096, 256, 0, stream>>>(mean_s, h1_t, edge_dst, edge_src, NE, HD);
    divide_mean<<<4096, 256, 0, stream>>>(mean_t, cnt_t, NTAXON, HD);
    divide_mean<<<4096, 256, 0, stream>>>(mean_s, cnt_s, NSOTU, HD);

    // ---- layer 2 GEMMs ----
    launch_gemm(mean_t, Wl2_st, nullptr, h2_t, NTAXON, HD, HD, 0, stream);
    launch_gemm(h1_t, Wr2_st, bl2_st, h2_t, NTAXON, HD, HD, F_ACCUM | F_RELU, stream);
    launch_gemm(mean_s, Wl2_ts, nullptr, h2_s, NSOTU, HD, HD, 0, stream);
    launch_gemm(h1_s, Wr2_ts, bl2_ts, h2_s, NSOTU, HD, HD, F_ACCUM | F_RELU, stream);

    // ---- per-node-type projection (writes into h1 buffers, now dead) ----
    launch_gemm(h2_s, Wlin_s, blin_s, z_s, NSOTU, HD, HD, 0, stream);
    launch_gemm(h2_t, Wlin_t, blin_t, z_t, NTAXON, HD, HD, 0, stream);

    // ---- edge decoder ----
    gather_concat<<<4096, 256, 0, stream>>>((float4*)zcat, (const float4*)z_s,
                                            (const float4*)z_t, label_src, label_dst,
                                            NEL, HD / 4);
    launch_gemm(zcat, Wd1, bd1, hdec, NEL, HD, 2 * HD, F_RELU, stream);
    decoder_out<<<4096, 256, 0, stream>>>((const float4*)hdec, (const float4*)Wd2,
                                          bd2, out, NEL, HD / 4);
}
